// AutoRegressiveGlobalSelfAttention_19576460935282
// MI455X (gfx1250) — matrix operand order repaired and checked by reading, not compiled, not run on hardware
//
#include <hip/hip_runtime.h>
#include <hip/hip_bf16.h>
#include <stdint.h>

// ---------------------------------------------------------------------------
// Problem constants (fixed by the reference): B=2, S=2048, D=1024, H=16, hd=64
// ---------------------------------------------------------------------------
constexpr int Bc  = 2;
constexpr int Sc  = 2048;
constexpr int Dc  = 1024;
constexpr int Hc  = 16;
constexpr int HDc = 64;
constexpr int Mg  = Bc * Sc;   // 4096 GEMM rows
constexpr int Kg  = Dc;        // 1024 GEMM reduction
constexpr int Ng  = Dc;        // 1024 GEMM cols

typedef __attribute__((ext_vector_type(8)))  _Float16 v8h;
typedef __attribute__((ext_vector_type(16))) _Float16 v16h;
typedef __attribute__((ext_vector_type(8)))  float    v8f;

__device__ inline v8f wmma_f16(v16h a, v16h b, v8f c) {
    // D = A(16x32 f16) * B(32x16 f16) + C(16x16 f32)
    return __builtin_amdgcn_wmma_f32_16x16x32_f16(
        /*neg_a=*/false, a, /*neg_b=*/false, b,
        /*c_mod=*/(short)0, c, /*reuse_a=*/false, /*reuse_b=*/false);
}

// Build a 16-half fragment from two 16-byte-aligned 8-half chunks.
__device__ inline v16h load_frag16(const _Float16* p0, const _Float16* p1) {
    union { v16h v; v8h h[2]; } u;
    u.h[0] = *(const v8h*)p0;
    u.h[1] = *(const v8h*)p1;
    return u.v;
}

// HBM -> LDS direct async copy (16B per lane), tracked by ASYNCcnt.
// LDS byte offset = low 32 bits of the generic pointer to __shared__ memory.
__device__ inline void async_load_b128(const void* gptr, void* lptr) {
    const uint32_t la = (uint32_t)(uintptr_t)lptr;
    asm volatile("global_load_async_to_lds_b128 %0, %1, off"
                 :: "v"(la), "v"(gptr)
                 : "memory");
}
__device__ inline void wait_async0() {
    asm volatile("s_wait_asynccnt 0x0" ::: "memory");
}

// ---------------------------------------------------------------------------
// Kernel: fp32 -> fp16 convert (grid-stride)
// ---------------------------------------------------------------------------
__global__ void cvt_f32_f16_kernel(const float* __restrict__ in,
                                   _Float16* __restrict__ out, int n) {
    int i = blockIdx.x * blockDim.x + threadIdx.x;
    int stride = gridDim.x * blockDim.x;
    for (; i < n; i += stride) out[i] = (_Float16)in[i];
}

// ---------------------------------------------------------------------------
// Kernel: out = X @ W^T  (X: [Mg,Kg] f16 row-major, W: [Ng,Kg] f16 row-major)
// MODE 0: write split-head  [B,H,S,hd] f16      (Q, K)
// MODE 1: write transposed  [B,H,hd,S] f16      (V^T, for the context GEMM)
// MODE 2: write plain f32 [Mg,Ng] + bias        (final output projection)
// Workgroup: 256 threads = 8 waves; tile 128(M) x 64(N); K chunk 32.
// Double-buffered HBM->LDS staging via global_load_async_to_lds_b128.
// Wave (wm,wn) in a 4x2 grid owns a 32x32 sub-tile -> 4 WMMA accumulators.
// ---------------------------------------------------------------------------
template <int MODE>
__global__ __launch_bounds__(256) void gemm_xwT_kernel(
        const _Float16* __restrict__ X,
        const _Float16* __restrict__ W,
        const float*    __restrict__ bias,
        void*           __restrict__ outp) {
    __shared__ _Float16 sA[2][128][40];   // 32-col K chunk, +8 halfs pad
    __shared__ _Float16 sB[2][64][40];

    const int tid  = threadIdx.x;
    const int lane = tid & 31;
    const int wv   = tid >> 5;
    const int wm   = wv >> 1;          // 0..3
    const int wn   = wv & 1;           // 0..1
    const int m0   = blockIdx.y * 128;
    const int n0   = blockIdx.x * 64;
    const int hsel = lane >> 4;        // which half of the wave
    const int l15  = lane & 15;

    // staging coordinates: 16B (8 halfs) per lane
    const int sr = tid >> 2;           // 0..63
    const int sc = (tid & 3) * 8;      // 0/8/16/24

    auto stage = [&](int buf, int k0) {
        async_load_b128(&X[(size_t)(m0 + sr)      * Kg + k0 + sc], &sA[buf][sr][sc]);
        async_load_b128(&X[(size_t)(m0 + sr + 64) * Kg + k0 + sc], &sA[buf][sr + 64][sc]);
        async_load_b128(&W[(size_t)(n0 + sr)      * Kg + k0 + sc], &sB[buf][sr][sc]);
    };

    v8f acc[2][2] = {};
    stage(0, 0);

    constexpr int NCHUNK = Kg / 32;    // 32
    for (int i = 0; i < NCHUNK; ++i) {
        wait_async0();                 // my async writes for chunk i landed
        __syncthreads();               // everyone's landed; prior reads done
        if (i + 1 < NCHUNK) stage((i + 1) & 1, (i + 1) * 32);
        const int buf = i & 1;

        v16h a[2], b[2];
        // A fragment: lane = M row; lanes 0-15 hold K {0..7,16..23}, 16-31 {8..15,24..31}
        for (int mt = 0; mt < 2; ++mt) {
            const _Float16* p = &sA[buf][wm * 32 + mt * 16 + l15][hsel * 8];
            a[mt] = load_frag16(p, p + 16);
        }
        // B fragment: lane = N col; lanes 0-15 hold K {0..7,16..23}, 16-31 {8..15,24..31} (W row-major == B^T)
        for (int nt = 0; nt < 2; ++nt) {
            const _Float16* p = &sB[buf][wn * 32 + nt * 16 + l15][hsel * 8];
            b[nt] = load_frag16(p, p + 16);
        }
        for (int mt = 0; mt < 2; ++mt)
            for (int nt = 0; nt < 2; ++nt)
                acc[mt][nt] = wmma_f16(a[mt], b[nt], acc[mt][nt]);
    }

    // C/D layout: VGPR v -> M = v (lanes 0-15) / v+8 (lanes 16-31); N = lane&15
    for (int mt = 0; mt < 2; ++mt)
        for (int nt = 0; nt < 2; ++nt)
            for (int v = 0; v < 8; ++v) {
                const int gm = m0 + wm * 32 + mt * 16 + v + hsel * 8;
                const int gn = n0 + wn * 32 + nt * 16 + l15;
                const float val = acc[mt][nt][v];
                if (MODE == 0) {
                    const int bb = gm / Sc, s = gm % Sc, h = gn / HDc, hd = gn % HDc;
                    ((_Float16*)outp)[(((size_t)bb * Hc + h) * Sc + s) * HDc + hd] =
                        (_Float16)val;
                } else if (MODE == 1) {
                    const int bb = gm / Sc, s = gm % Sc, h = gn / HDc, hd = gn % HDc;
                    ((_Float16*)outp)[(((size_t)bb * Hc + h) * HDc + hd) * Sc + s] =
                        (_Float16)val;
                } else {
                    ((float*)outp)[(size_t)gm * Ng + gn] = val + bias[gn];
                }
            }
}

// ---------------------------------------------------------------------------
// Kernel: attention core. One workgroup per (b*H+h, 16-query-row strip).
// Phase 1: scores S[16][2048] = Q16x64 . K^T via WMMA, causal+pad mask -> LDS
//          (mask bits pre-hoisted into registers -> branch-free v_cndmask)
// Phase 2: row softmax (shuffle reductions), write proba (fp32, global) once,
//          keep f16 copy of P in LDS
// Phase 3: context C[16][64] = P @ V via WMMA (V pre-transposed [B,H,hd,S])
// ---------------------------------------------------------------------------
__global__ __launch_bounds__(256) void attn_kernel(
        const _Float16*      __restrict__ Q,    // [B,H,S,hd]
        const _Float16*      __restrict__ Kb,   // [B,H,S,hd]
        const _Float16*      __restrict__ VT,   // [B,H,hd,S]
        const unsigned char* __restrict__ mask, // [B,S] bool
        float*               __restrict__ proba, // [B,H,S,S]
        _Float16*            __restrict__ ctx)   // [B,S,D]
{
    __shared__ float    sS[16][2064];   // 132,096 B  (stride pad keeps 16B align)
    __shared__ _Float16 sP[16][2064];   //  66,048 B

    const int tid  = threadIdx.x;
    const int lane = tid & 31;
    const int wv   = tid >> 5;
    const int hsel = lane >> 4;
    const int l15  = lane & 15;
    const int qt   = blockIdx.x;        // 0..127
    const int bh   = blockIdx.y;        // 0..31
    const int bb   = bh >> 4;
    const int h    = bh & 15;
    const int q0   = qt * 16;
    constexpr float NEG = -3.402823466e38f;

    const unsigned char* mrow = mask + bb * Sc;

    // ---- Phase 1: scores -------------------------------------------------
    // Hoist this thread's 8 query-mask bits into a register bitfield.
    unsigned mqbits = 0;
    for (int v = 0; v < 8; ++v)
        mqbits |= (unsigned)(mrow[q0 + v + hsel * 8] != 0) << v;

    const _Float16* Qp = Q + ((size_t)bh * Sc + q0) * HDc;
    v16h afrag[2];
    for (int c2 = 0; c2 < 2; ++c2) {
        const _Float16* p = Qp + (size_t)l15 * HDc + c2 * 32 + hsel * 8;
        afrag[c2] = load_frag16(p, p + 16);
    }
    for (int kt = wv; kt < Sc / 16; kt += 8) {
        const int kc = kt * 16 + l15;
        const unsigned mk = (unsigned)(mrow[kc] != 0);   // one u8 load per tile
        const _Float16* Kp = Kb + ((size_t)bh * Sc + kt * 16) * HDc;
        v8f c = {};
        for (int c2 = 0; c2 < 2; ++c2) {
            const _Float16* p = Kp + (size_t)l15 * HDc + c2 * 32 + hsel * 8;
            v16h bfrag = load_frag16(p, p + 16);
            c = wmma_f16(afrag[c2], bfrag, c);
        }
        for (int v = 0; v < 8; ++v) {
            const int qr = v + hsel * 8;
            const unsigned valid =
                (unsigned)(kc <= q0 + qr) & mk & ((mqbits >> v) & 1u);
            sS[qr][kc] = valid ? c[v] : NEG;
        }
    }
    __syncthreads();

    // ---- Phase 2: softmax + proba write ---------------------------------
    {
        const int row = tid >> 4;       // 16 threads per row
        const int ts  = tid & 15;
        float m = NEG;
        for (int c = ts; c < Sc; c += 16) m = fmaxf(m, sS[row][c]);
        for (int off = 8; off; off >>= 1) m = fmaxf(m, __shfl_xor(m, off, 16));
        float sum = 0.f;
        for (int c = ts; c < Sc; c += 16) {
            const float e = __expf(sS[row][c] - m);
            sS[row][c] = e;
            sum += e;
        }
        for (int off = 8; off; off >>= 1) sum += __shfl_xor(sum, off, 16);
        const float inv = 1.0f / sum;
        float* prow = proba + ((size_t)bh * Sc + (q0 + row)) * Sc;
        for (int c = ts; c < Sc; c += 16) {
            const float p = sS[row][c] * inv;
            prow[c]    = p;
            sP[row][c] = (_Float16)p;
        }
    }
    __syncthreads();

    // ---- Phase 3: context = P @ V ---------------------------------------
    if (wv < 4) {
        const int nt = wv;              // 16-wide hd tile
        const _Float16* Vp = VT + ((size_t)bh * HDc + nt * 16) * Sc;
        v8f c = {};
        for (int kc = 0; kc < Sc / 32; ++kc) {
            const _Float16* pa = &sP[l15][kc * 32 + hsel * 8];
            v16h af = load_frag16(pa, pa + 16);
            const _Float16* pb = Vp + (size_t)l15 * Sc + kc * 32 + hsel * 8;
            v16h bf = load_frag16(pb, pb + 16);
            c = wmma_f16(af, bf, c);
        }
        for (int v = 0; v < 8; ++v) {
            const int qr  = v + hsel * 8;
            const int dim = nt * 16 + l15;
            ctx[((size_t)bb * Sc + q0 + qr) * Dc + h * HDc + dim] = (_Float16)c[v];
        }
    }
}

// ---------------------------------------------------------------------------
// Host launcher
// ---------------------------------------------------------------------------
extern "C" void kernel_launch(void* const* d_in, const int* in_sizes, int n_in,
                              void* d_out, int out_size, void* d_ws, size_t ws_size,
                              hipStream_t stream) {
    const float*         hs   = (const float*)d_in[0];         // [B,S,D]
    const unsigned char* mask = (const unsigned char*)d_in[1]; // [B,S] bool
    const float*         Wq   = (const float*)d_in[2];
    const float*         Wk   = (const float*)d_in[3];
    const float*         Wv   = (const float*)d_in[4];
    const float*         Wo   = (const float*)d_in[5];
    const float*         bo   = (const float*)d_in[6];

    float* out   = (float*)d_out;                       // [B,S,D]
    float* proba = out + (size_t)Bc * Sc * Dc;          // [B,H,S,S]

    // Workspace layout (f16 buffers), ~48 MB total
    char* ws = (char*)d_ws;
    constexpr size_t SZ_X  = (size_t)Mg * Kg * 2;       // 8 MB
    constexpr size_t SZ_W  = (size_t)Ng * Kg * 2;       // 2 MB
    constexpr size_t SZ_QK = (size_t)Bc * Sc * Dc * 2;  // 8 MB
    _Float16* Xh  = (_Float16*)(ws);
    _Float16* Wqh = (_Float16*)(ws + SZ_X);
    _Float16* Wkh = (_Float16*)(ws + SZ_X + SZ_W);
    _Float16* Wvh = (_Float16*)(ws + SZ_X + 2 * SZ_W);
    _Float16* Woh = (_Float16*)(ws + SZ_X + 3 * SZ_W);
    _Float16* Qb  = (_Float16*)(ws + SZ_X + 4 * SZ_W);
    _Float16* Kbf = (_Float16*)(ws + SZ_X + 4 * SZ_W + SZ_QK);
    _Float16* VTb = (_Float16*)(ws + SZ_X + 4 * SZ_W + 2 * SZ_QK);
    _Float16* Ctx = (_Float16*)(ws + SZ_X + 4 * SZ_W + 3 * SZ_QK);

    // 1) fp32 -> fp16 converts
    cvt_f32_f16_kernel<<<1024, 256, 0, stream>>>(hs, Xh, Mg * Kg);
    cvt_f32_f16_kernel<<<256, 256, 0, stream>>>(Wq, Wqh, Ng * Kg);
    cvt_f32_f16_kernel<<<256, 256, 0, stream>>>(Wk, Wkh, Ng * Kg);
    cvt_f32_f16_kernel<<<256, 256, 0, stream>>>(Wv, Wvh, Ng * Kg);
    cvt_f32_f16_kernel<<<256, 256, 0, stream>>>(Wo, Woh, Ng * Kg);

    // 2) QKV projections (WMMA GEMM): Q,K split-head; V pre-transposed
    dim3 ggrid(Ng / 64, Mg / 128);   // 16 x 32
    gemm_xwT_kernel<0><<<ggrid, 256, 0, stream>>>(Xh, Wqh, nullptr, (void*)Qb);
    gemm_xwT_kernel<0><<<ggrid, 256, 0, stream>>>(Xh, Wkh, nullptr, (void*)Kbf);
    gemm_xwT_kernel<1><<<ggrid, 256, 0, stream>>>(Xh, Wvh, nullptr, (void*)VTb);

    // 3) attention core: scores -> softmax -> proba + context
    dim3 agrid(Sc / 16, Bc * Hc);    // 128 x 32
    attn_kernel<<<agrid, 256, 0, stream>>>(Qb, Kbf, VTb, mask, proba, Ctx);

    // 4) output projection with bias (fp32 out)
    gemm_xwT_kernel<2><<<ggrid, 256, 0, stream>>>(Ctx, Woh, bo, (void*)out);
}
